// ModelLSTM_1984274890795
// MI455X (gfx1250) — compile-verified, matrix operand order checked by reading, not run on hardware
//
#include <hip/hip_runtime.h>
#include <hip/hip_bf16.h>

#define HIDN   512
#define EMBD   511
#define NSTEP  36
#define BATCH  16384
#define KTOT   1024          // 512 (emb|y) + 512 (h)
#define TILEB  32            // batch rows per workgroup (2 WMMA m-tiles)
#define NTHREADS 512         // 16 wave32

typedef __attribute__((ext_vector_type(16))) __bf16 bf16x16;
typedef __attribute__((ext_vector_type(8)))  __bf16 bf16x8;
typedef __attribute__((ext_vector_type(8)))  float  f32x8;

__device__ __forceinline__ float fast_sigmoid(float x) {
  return 1.0f / (1.0f + __expf(-x));
}
__device__ __forceinline__ float fast_tanh(float x) {
  x = fminf(fmaxf(x, -15.0f), 15.0f);
  float e = __expf(-2.0f * x);
  return (1.0f - e) / (1.0f + e);
}

// ---------------------------------------------------------------------------
// Pack [Wih | Whh] (2048 x 1024 fp32, row = gate output n, col = k) into the
// WMMA B-operand (32x16 bf16, K x N) lane layout, blocked as
//   block id = ((gate*32 + n_tile)*32 + k_chunk), 1 KB per block,
//   lane*32B holds its 16 bf16 values (t<8 -> K=kbase+t, t>=8 -> K=kbase+16+t-8,
//   kbase = lane<16 ? 0 : 8, N = lane&15).
// ---------------------------------------------------------------------------
__global__ void pack_weights(const float* __restrict__ Wih,
                             const float* __restrict__ Whh,
                             __bf16* __restrict__ packed) {
  int e = blockIdx.x * blockDim.x + threadIdx.x;   // 0 .. 2048*1024-1
  if (e >= 2048 * 1024) return;
  int bid  = e >> 9;          // 512 elements per 1KB block
  int r    = e & 511;
  int lane = r >> 4;
  int t    = r & 15;
  int kk = bid & 31;
  int jt = (bid >> 5) & 31;
  int g  = bid >> 10;
  int n  = g * 512 + jt * 16 + (lane & 15);
  int kbase = (lane < 16) ? 0 : 8;
  int K = (t < 8) ? (kbase + t) : (kbase + 16 + (t - 8));
  int k = kk * 32 + K;
  float v = (k < 512) ? Wih[n * 512 + k] : Whh[n * 512 + (k - 512)];
  packed[e] = (__bf16)v;
}

// ---------------------------------------------------------------------------
// emb = relu(x @ W1^T + b1) -> bf16 panel [B,512]; col 511 is the y slot
// (filled per-workgroup in LDS by the LSTM kernel). K=23 is tiny.
// ---------------------------------------------------------------------------
__global__ void embed_relu(const float* __restrict__ x,
                           const float* __restrict__ W1,
                           const float* __restrict__ b1,
                           __bf16* __restrict__ inpG) {
  int idx = blockIdx.x * blockDim.x + threadIdx.x;  // B*512
  if (idx >= BATCH * 512) return;
  int row = idx >> 9;
  int col = idx & 511;
  float v = 0.0f;
  if (col < EMBD) {
    float acc = b1[col];
#pragma unroll
    for (int k = 0; k < 23; ++k)
      acc = fmaf(x[row * 23 + k], W1[col * 23 + k], acc);
    v = fmaxf(acc, 0.0f);
  }
  inpG[idx] = (__bf16)v;
}

// ---------------------------------------------------------------------------
// Persistent LSTM: one workgroup owns TILEB batch rows for all 36 steps.
// LDS holds the bf16 A panel [32][1024] = 64KB: cols 0..510 emb, 511 y,
// 512..1023 h.  c stays in VGPRs.  Wave w handles hidden-column quads
// {2w, 2w+1}; for each quad it computes all 4 gate tiles for both m-tiles.
// ---------------------------------------------------------------------------
__launch_bounds__(NTHREADS, 1)
__global__ void lstm_persist(const __bf16* __restrict__ inpG,
                             const __bf16* __restrict__ packedW,
                             const float* __restrict__ bih,
                             const float* __restrict__ bhh,
                             const float* __restrict__ W3,
                             const float* __restrict__ b3,
                             float* __restrict__ out) {
  __shared__ __align__(16) __bf16 Abuf[TILEB][KTOT];   // 65536 B

  const int tid    = threadIdx.x;
  const int w      = tid >> 5;
  const int lane   = tid & 31;
  const int laneN  = lane & 15;
  const int hiHalf = lane >> 4;
  const int row0   = blockIdx.x * TILEB;
  const float b3v  = b3[0];

  // ---- stage emb tile into LDS, zero h section -----------------------------
  {
    const bf16x8* src = (const bf16x8*)(inpG + (size_t)row0 * 512);
    bf16x8 z;
#pragma unroll
    for (int r = 0; r < 8; ++r) z[r] = (__bf16)0.0f;
    for (int i = tid; i < TILEB * 512 / 8; i += NTHREADS) {
      int rr = i >> 6;              // 64 8-elem chunks per row
      int cc = i & 63;
      *(bf16x8*)&Abuf[rr][cc * 8]       = src[(size_t)rr * 64 + cc];
      *(bf16x8*)&Abuf[rr][512 + cc * 8] = z;
    }
  }
  if (tid < TILEB) {
    Abuf[tid][511] = (__bf16)(-0.8f);                 // x0 = -0.8
    out[row0 + tid] = b3v;                            // step 0 y accumulator
  }
  __syncthreads();

  // ---- persistent per-wave state ------------------------------------------
  float cst[2][2][8];
#pragma unroll
  for (int q = 0; q < 2; ++q)
#pragma unroll
    for (int m = 0; m < 2; ++m)
#pragma unroll
      for (int r = 0; r < 8; ++r) cst[q][m][r] = 0.0f;

  float biasv[2][4], w3v[2];
#pragma unroll
  for (int q = 0; q < 2; ++q) {
    int j = (w * 2 + q) * 16;
#pragma unroll
    for (int g = 0; g < 4; ++g)
      biasv[q][g] = bih[g * 512 + j + laneN] + bhh[g * 512 + j + laneN];
    w3v[q] = W3[j + laneN];
  }

  const __bf16* bp = packedW + (size_t)lane * 16;     // lane slot inside block

  for (int step = 0; step < NSTEP; ++step) {
    f32x8 acc[2][4][2];                               // [quad][gate][mtile]

    // ---- phase 1: gates = [inp|h] @ [Wih|Whh]^T + bias (WMMA bf16) --------
#pragma unroll
    for (int q = 0; q < 2; ++q) {
      const int jt = w * 2 + q;
#pragma unroll
      for (int g = 0; g < 4; ++g)
#pragma unroll
        for (int m = 0; m < 2; ++m)
#pragma unroll
          for (int r = 0; r < 8; ++r) acc[q][g][m][r] = biasv[q][g];

#pragma unroll 1
      for (int kk = 0; kk < 32; ++kk) {
        bf16x16 a[2];
        const int kofs = kk * 32 + hiHalf * 8;
#pragma unroll
        for (int m = 0; m < 2; ++m) {
          union { bf16x16 v; bf16x8 h[2]; } u;
          u.h[0] = *(const bf16x8*)&Abuf[m * 16 + laneN][kofs];
          u.h[1] = *(const bf16x8*)&Abuf[m * 16 + laneN][kofs + 16];
          a[m] = u.v;
        }
        bf16x16 bw[4];
#pragma unroll
        for (int g = 0; g < 4; ++g) {
          size_t blk = ((size_t)(g * 32 + jt) * 32 + kk) * 512;
          bw[g] = *(const bf16x16*)(bp + blk);
          __builtin_prefetch((const void*)(bp + blk + 512), 0, 1);
        }
#pragma unroll
        for (int g = 0; g < 4; ++g)
#pragma unroll
          for (int m = 0; m < 2; ++m)
            acc[q][g][m] = __builtin_amdgcn_wmma_f32_16x16x32_bf16(
                false, a[m], false, bw[g], (short)0, acc[q][g][m],
                false, false);
      }
    }

    __syncthreads();   // all reads of old h/y done before h is overwritten

    // ---- phase 2: LSTM cell (VALU), h -> LDS bf16, y partial -> atomic ----
#pragma unroll
    for (int q = 0; q < 2; ++q) {
      const int j = (w * 2 + q) * 16;
#pragma unroll
      for (int m = 0; m < 2; ++m) {
#pragma unroll
        for (int r = 0; r < 8; ++r) {
          float iv = fast_sigmoid(acc[q][0][m][r]);
          float fv = fast_sigmoid(acc[q][1][m][r]);
          float gv = fast_tanh   (acc[q][2][m][r]);
          float ov = fast_sigmoid(acc[q][3][m][r]);
          float cn = fmaf(fv, cst[q][m][r], iv * gv);
          cst[q][m][r] = cn;
          float h = ov * fast_tanh(cn);
          int row = m * 16 + r + hiHalf * 8;
          Abuf[row][512 + j + laneN] = (__bf16)h;
          // y = h @ W3^T : reduce the 16 lanes sharing this row
          float v = h * w3v[q];
          v += __shfl_xor(v, 1);
          v += __shfl_xor(v, 2);
          v += __shfl_xor(v, 4);
          v += __shfl_xor(v, 8);
          if (laneN == 0)
            atomicAdd(&out[(size_t)step * BATCH + row0 + row], v);
        }
      }
    }

    __threadfence();
    __syncthreads();
    if (tid < TILEB) {
      float y = out[(size_t)step * BATCH + row0 + tid];   // b3 + all partials
      Abuf[tid][511] = (__bf16)y;                         // feed back
      if (step + 1 < NSTEP)
        out[(size_t)(step + 1) * BATCH + row0 + tid] = b3v;
    }
    __syncthreads();
  }
}

// ---------------------------------------------------------------------------
extern "C" void kernel_launch(void* const* d_in, const int* in_sizes, int n_in,
                              void* d_out, int out_size, void* d_ws, size_t ws_size,
                              hipStream_t stream) {
  (void)in_sizes; (void)n_in; (void)out_size; (void)ws_size;
  const float* x   = (const float*)d_in[0];
  const float* W1  = (const float*)d_in[1];
  const float* b1  = (const float*)d_in[2];
  const float* Wih = (const float*)d_in[3];
  const float* bih = (const float*)d_in[4];
  const float* Whh = (const float*)d_in[5];
  const float* bhh = (const float*)d_in[6];
  const float* W3  = (const float*)d_in[7];
  const float* b3  = (const float*)d_in[8];
  float* out = (float*)d_out;

  __bf16* inpG    = (__bf16*)d_ws;                                  // 16 MiB
  __bf16* packedW = (__bf16*)((char*)d_ws + (size_t)BATCH * 512 * 2); // 4 MiB

  pack_weights<<<(2048 * 1024) / 256, 256, 0, stream>>>(Wih, Whh, packedW);
  embed_relu<<<(BATCH * 512) / 256, 256, 0, stream>>>(x, W1, b1, inpG);
  lstm_persist<<<BATCH / TILEB, NTHREADS, 0, stream>>>(
      inpG, packedW, bih, bhh, W3, b3, out);
}
